// RetMIL_60309930770640
// MI455X (gfx1250) — compile-verified, matrix operand order checked by reading, not run on hardware
//
#include <hip/hip_runtime.h>
#include <hip/hip_bf16.h>
#include <math.h>

typedef __bf16 bf16_t;
typedef __attribute__((ext_vector_type(16))) __bf16 v16bf;
typedef __attribute__((ext_vector_type(8)))  __bf16 v8bf;
typedef __attribute__((ext_vector_type(8)))  float  v8f;
typedef int i32x4 __attribute__((vector_size(4 * sizeof(int))));

#define EMBED 512
#define HEADS 8
#define DH    64
#define NTOK  16384
#define TLOC  512
#define NCH   32
#define TGLB  32
#define DPOOL 256
#define KTILE 32

#if __has_builtin(__builtin_amdgcn_global_load_async_to_lds_b128) && \
    __has_builtin(__builtin_amdgcn_s_wait_asynccnt)
#define HAVE_ASYNC_LDS 1
#else
#define HAVE_ASYNC_LDS 0
#endif

// ---------------------------------------------------------------------------
// Fragment helpers (CDNA5 wave32 WMMA 16x16x32 bf16 layouts, CDNA5 ISA)
// A (16xK-row-major) / B-from-[N][K]: lane&15 -> row, (lane>>4)*8 -> K-chunk base,
// two 16B chunks at k+kb and k+kb+16.
// C/D: lane&15 -> N col, vgpr r -> M = r + 8*(lane>=16).
// ---------------------------------------------------------------------------
__device__ __forceinline__ v16bf load_frag(const bf16_t* p, int ld, int k0) {
    const int lane = threadIdx.x & 31;
    const int row  = lane & 15;
    const int kb   = (lane >> 4) << 3;
    const bf16_t* base = p + row * ld + k0 + kb;
    v8bf lo = *reinterpret_cast<const v8bf*>(base);
    v8bf hi = *reinterpret_cast<const v8bf*>(base + 16);
    v16bf f;
#pragma unroll
    for (int i = 0; i < 8; ++i) { f[i] = lo[i]; f[i + 8] = hi[i]; }
    return f;
}

__device__ __forceinline__ v8f wmma_bf16(v16bf a, v16bf b, v8f c) {
    return __builtin_amdgcn_wmma_f32_16x16x32_bf16(false, a, false, b, (short)0, c,
                                                   false, false);
}

// ---------------------------------------------------------------------------
// f32 -> bf16 convert
// ---------------------------------------------------------------------------
__global__ void k_cvt_bf16(const float* __restrict__ in, bf16_t* __restrict__ out, long n) {
    long i = (long)blockIdx.x * blockDim.x + threadIdx.x;
    if (i < n) out[i] = (bf16_t)in[i];
}

// ---------------------------------------------------------------------------
// Stage one 64xKTILE bf16 B-tile into LDS, distributed over 128 threads
// (2 x 16B chunks per thread). Async-to-LDS when the toolchain exposes it.
// ---------------------------------------------------------------------------
__device__ __forceinline__ void stage_btile(const bf16_t* __restrict__ B, int K,
                                            int n0, int k0, bf16_t* dst) {
    const int tid = threadIdx.x;
#pragma unroll
    for (int i = 0; i < 2; ++i) {
        int chunk = tid + i * 128;            // 0..255 : 64 rows x 4 chunks
        int row = chunk >> 2;
        int col = (chunk & 3) << 3;           // halves
        const bf16_t* g = B + (long)(n0 + row) * K + k0 + col;
        bf16_t* l = dst + row * KTILE + col;
#if HAVE_ASYNC_LDS
        __builtin_amdgcn_global_load_async_to_lds_b128(
            (__attribute__((address_space(1))) i32x4*)(void*)g,
            (__attribute__((address_space(3))) i32x4*)(void*)l, 0, 0);
#else
        *reinterpret_cast<v8bf*>(l) = *reinterpret_cast<const v8bf*>(g);
#endif
    }
}

// ---------------------------------------------------------------------------
// GEMM: C[MxN] = A[MxK](bf16,row) * B[NxK](bf16,row)^T (+bias).
// Block = 4 waves, each wave a 16x64 tile; B tile double-buffered through LDS
// via async-to-LDS so the next K-slab streams in under the current WMMAs.
// ---------------------------------------------------------------------------
__global__ void k_gemm(const bf16_t* __restrict__ A, const bf16_t* __restrict__ B,
                       const float* __restrict__ bias,
                       float* __restrict__ Cf, bf16_t* __restrict__ Cb,
                       int M, int N, int K) {
    __shared__ __align__(16) bf16_t bt[2][64 * KTILE];
    const int wave = threadIdx.x >> 5;
    const int lane = threadIdx.x & 31;
    const int mt = blockIdx.y * 4 + wave;
    const bool active = (mt * 16 < M);
    const int mtc = active ? mt : 0;          // clamp: keep all waves in barriers
    const int n0 = blockIdx.x * 64;
    const bf16_t* Ap = A + (long)mtc * 16 * K;

    v8f acc[4] = {{}, {}, {}, {}};
    const int nk = K / KTILE;
    stage_btile(B, K, n0, 0, bt[0]);
    for (int kt = 0; kt < nk; ++kt) {
        const int cur = kt & 1;
        if (kt + 1 < nk) {
            stage_btile(B, K, n0, (kt + 1) * KTILE, bt[cur ^ 1]);
#if HAVE_ASYNC_LDS
            __builtin_amdgcn_s_wait_asynccnt(2);   // tile 'cur' complete, next in flight
#endif
        } else {
#if HAVE_ASYNC_LDS
            __builtin_amdgcn_s_wait_asynccnt(0);
#endif
        }
        __syncthreads();
        v16bf a = load_frag(Ap, K, kt * KTILE);
#pragma unroll
        for (int j = 0; j < 4; ++j) {
            v16bf b = load_frag(bt[cur] + j * 16 * KTILE, KTILE, 0);
            acc[j] = wmma_bf16(a, b, acc[j]);
        }
        __syncthreads();
    }

    if (!active) return;
    const int n  = lane & 15;
    const int mb = (lane >> 4) << 3;
#pragma unroll
    for (int j = 0; j < 4; ++j) {
        int ncol = n0 + j * 16 + n;
        float badd = bias ? bias[ncol] : 0.f;
#pragma unroll
        for (int r = 0; r < 8; ++r) {
            float v = acc[j][r] + badd;
            long idx = (long)(mt * 16 + mb + r) * N + ncol;
            if (Cf) Cf[idx] = v;
            if (Cb) Cb[idx] = (bf16_t)v;
        }
    }
}

// ---------------------------------------------------------------------------
// RoPE prep: rotary on q,k (k pre-scaled), repack head-major; v -> transposed DhxT.
// ---------------------------------------------------------------------------
__global__ void k_rope_prep(const bf16_t* __restrict__ q, const bf16_t* __restrict__ k,
                            const bf16_t* __restrict__ v,
                            bf16_t* __restrict__ qr, bf16_t* __restrict__ kr,
                            bf16_t* __restrict__ vT, int T, int total_pairs, float kscale) {
    int idx = blockIdx.x * blockDim.x + threadIdx.x;
    if (idx >= total_pairs) return;
    int cp  = idx % (EMBED / 2);
    int row = idx / (EMBED / 2);
    int c = row / T, t = row % T;
    int ch = cp * 2;
    int h = ch / DH, dh = ch % DH;
    int i = dh >> 1;
    float ang = __expf(-(float)i * (logf(10000.f) / (float)(DH / 2 - 1)));
    float sv, cv;
    __sincosf((float)t * ang, &sv, &cv);
    long in0 = (long)row * EMBED + ch;
    float q0 = (float)q[in0], q1 = (float)q[in0 + 1];
    float k0 = (float)k[in0] * kscale, k1 = (float)k[in0 + 1] * kscale;
    long ho = ((long)c * HEADS + h) * T;
    long o0 = (ho + t) * DH + dh;
    qr[o0]     = (bf16_t)(q0 * cv - q1 * sv);
    qr[o0 + 1] = (bf16_t)(q1 * cv + q0 * sv);
    kr[o0]     = (bf16_t)(k0 * cv - k1 * sv);
    kr[o0 + 1] = (bf16_t)(k1 * cv + k0 * sv);
    long vo = ((long)c * HEADS + h) * DH;
    vT[(vo + dh) * T + t]     = v[in0];
    vT[(vo + dh + 1) * T + t] = v[in0 + 1];
}

// ---------------------------------------------------------------------------
// Retention attention: wave = 16 query rows, stream 32-wide key tiles.
// qk WMMA -> decay mask + |.| denom accum -> LDS restage -> out += qk*v WMMA.
// ---------------------------------------------------------------------------
__global__ void k_retention(const bf16_t* __restrict__ qr, const bf16_t* __restrict__ kr,
                            const bf16_t* __restrict__ vT, float* __restrict__ att, int T) {
    __shared__ __align__(16) bf16_t lds[4][16 * 40];
    const int wave = threadIdx.x >> 5;
    const int lane = threadIdx.x & 31;
    const int h = blockIdx.y;
    const int c = blockIdx.z;
    const int t0 = (blockIdx.x * 4 + wave) * 16;
    if (t0 >= T) return;

    const float d = logf(1.f - exp2f(-5.f - (float)h));
    const float ed = __expf(d);
    const int n  = lane & 15;
    const int mb = (lane >> 4) << 3;

    float invS[8];
#pragma unroll
    for (int r = 0; r < 8; ++r) {
        int t = t0 + mb + r;
        float St = (1.f - __expf(d * (float)(t + 1))) / (1.f - ed);
        invS[r] = rsqrtf(St);
    }

    const long hc = (long)c * HEADS + h;
    const bf16_t* qbase = qr + hc * T * DH + (long)t0 * DH;
    const bf16_t* kbase = kr + hc * T * DH;
    const bf16_t* vbase = vT + hc * DH * T;

    v16bf a0 = load_frag(qbase, DH, 0);
    v16bf a1 = load_frag(qbase, DH, 32);

    v8f outacc[4] = {{}, {}, {}, {}};
    float dacc[8] = {0, 0, 0, 0, 0, 0, 0, 0};
    bf16_t* myl = lds[wave];

    const int smax = (t0 + 15) >> 5;
    for (int st = 0; st <= smax; ++st) {
        const int s0 = st * 32;
#pragma unroll
        for (int sub = 0; sub < 2; ++sub) {
            v8f qk = {};
            const bf16_t* kb2 = kbase + (long)(s0 + sub * 16) * DH;
            qk = wmma_bf16(a0, load_frag(kb2, DH, 0), qk);
            qk = wmma_bf16(a1, load_frag(kb2, DH, 32), qk);
#pragma unroll
            for (int r = 0; r < 8; ++r) {
                int t = t0 + mb + r;
                int s = s0 + sub * 16 + n;
                float m = (s <= t) ? __expf(d * (float)(t - s)) * invS[r] : 0.f;
                float v = qk[r] * m;
                dacc[r] += fabsf(v);
                myl[(mb + r) * 40 + sub * 16 + n] = (bf16_t)v;
            }
        }
        __asm volatile("s_wait_dscnt 0" ::: "memory");   // cross-lane LDS visibility
        v16bf afrag = load_frag(myl, 40, 0);
#pragma unroll
        for (int j = 0; j < 4; ++j)
            outacc[j] = wmma_bf16(afrag, load_frag(vbase + (long)(j * 16) * T, T, s0),
                                  outacc[j]);
        __asm volatile("s_wait_dscnt 0" ::: "memory");   // WAR before next-tile stores
    }

#pragma unroll
    for (int r = 0; r < 8; ++r) {
        float v = dacc[r];
        v += __shfl_xor(v, 1);
        v += __shfl_xor(v, 2);
        v += __shfl_xor(v, 4);
        v += __shfl_xor(v, 8);            // stays inside 16-lane half (wave32)
        dacc[r] = 1.f / fminf(fmaxf(v, 1.f), 50000.f);
    }
#pragma unroll
    for (int j = 0; j < 4; ++j)
#pragma unroll
        for (int r = 0; r < 8; ++r) {
            int t = t0 + mb + r;
            att[((long)c * T + t) * EMBED + h * DH + j * 16 + n] = outacc[j][r] * dacc[r];
        }
}

// ---------------------------------------------------------------------------
// Per-head RMS (Dh=64) + SiLU(g) gate. Writes f32 and bf16 copies.
// ---------------------------------------------------------------------------
__global__ void k_rms_silu(const float* __restrict__ att, const bf16_t* __restrict__ g,
                           float* __restrict__ loc, bf16_t* __restrict__ locb, int rows) {
    int idx = blockIdx.x * blockDim.x + threadIdx.x;
    if (idx >= rows * HEADS) return;
    int row = idx / HEADS, h = idx % HEADS;
    long base = (long)row * EMBED + h * DH;
    float ss = 0.f;
    for (int dh = 0; dh < DH; ++dh) { float v = att[base + dh]; ss += v * v; }
    float sc = rsqrtf(ss * (1.f / DH) + 1e-6f);
    for (int dh = 0; dh < DH; ++dh) {
        float v  = att[base + dh] * sc;
        float gv = (float)g[base + dh];
        float o  = (gv / (1.f + __expf(-gv))) * v;
        loc[base + dh]  = o;
        locb[base + dh] = (bf16_t)o;
    }
}

// ---------------------------------------------------------------------------
// Pool scores: score[row] = dot(tanh(a)*sigmoid(b), wc) + bc
// ---------------------------------------------------------------------------
__global__ void k_pool_scores(const float* __restrict__ ap, const float* __restrict__ bp,
                              const float* __restrict__ wc, const float* __restrict__ bc,
                              float* __restrict__ scores, int rows) {
    int wave = threadIdx.x >> 5, lane = threadIdx.x & 31;
    int row = blockIdx.x * 4 + wave;
    if (row >= rows) return;
    float acc = 0.f;
    for (int i = lane; i < DPOOL; i += 32) {
        float a = tanhf(ap[(long)row * DPOOL + i]);
        float b = 1.f / (1.f + __expf(-bp[(long)row * DPOOL + i]));
        acc += a * b * wc[i];
    }
    for (int off = 16; off; off >>= 1) acc += __shfl_xor(acc, off);
    if (lane == 0) scores[row] = acc + bc[0];
}

// ---------------------------------------------------------------------------
// Softmax over T + weighted sum of x -> fe[blockIdx.x][EMBED]
// ---------------------------------------------------------------------------
__global__ void k_pool_weighted(const float* __restrict__ scores, const float* __restrict__ x,
                                float* __restrict__ fe, int T) {
    __shared__ float red[256];
    __shared__ float s_max, s_sum;
    int c = blockIdx.x, tid = threadIdx.x;
    const float* sc = scores + (long)c * T;
    float mx = -1e30f;
    for (int t = tid; t < T; t += 256) mx = fmaxf(mx, sc[t]);
    red[tid] = mx; __syncthreads();
    for (int s = 128; s; s >>= 1) { if (tid < s) red[tid] = fmaxf(red[tid], red[tid + s]); __syncthreads(); }
    if (tid == 0) s_max = red[0];
    __syncthreads();
    float sm = 0.f;
    for (int t = tid; t < T; t += 256) sm += __expf(sc[t] - s_max);
    red[tid] = sm; __syncthreads();
    for (int s = 128; s; s >>= 1) { if (tid < s) red[tid] += red[tid + s]; __syncthreads(); }
    if (tid == 0) s_sum = red[0];
    __syncthreads();
    float inv = 1.f / s_sum;
    for (int ch = tid; ch < EMBED; ch += 256) {
        float acc = 0.f;
        for (int t = 0; t < T; ++t)
            acc += __expf(sc[t] - s_max) * inv * x[((long)c * T + t) * EMBED + ch];
        fe[(long)c * EMBED + ch] = acc;
    }
}

// ---------------------------------------------------------------------------
// Classifier: out[j] = dot(fe, cls_w[j])
// ---------------------------------------------------------------------------
__global__ void k_classify(const float* __restrict__ fe, const float* __restrict__ w,
                           float* __restrict__ out) {
    int wave = threadIdx.x >> 5, lane = threadIdx.x & 31;
    if (wave >= 2) return;
    float acc = 0.f;
    for (int i = lane; i < EMBED; i += 32) acc += fe[i] * w[wave * EMBED + i];
    for (int off = 16; off; off >>= 1) acc += __shfl_xor(acc, off);
    if (lane == 0) out[wave] = acc;
}

// ---------------------------------------------------------------------------
extern "C" void kernel_launch(void* const* d_in, const int* in_sizes, int n_in,
                              void* d_out, int out_size, void* d_ws, size_t ws_size,
                              hipStream_t stream) {
    (void)in_sizes; (void)n_in; (void)out_size; (void)ws_size;
    const float* x    = (const float*)d_in[0];
    const float* lq   = (const float*)d_in[1];
    const float* lk   = (const float*)d_in[2];
    const float* lv   = (const float*)d_in[3];
    const float* lg   = (const float*)d_in[4];
    const float* la_w = (const float*)d_in[5];
    const float* la_b = (const float*)d_in[6];
    const float* lb_w = (const float*)d_in[7];
    const float* lb_b = (const float*)d_in[8];
    const float* lc_w = (const float*)d_in[9];
    const float* lc_b = (const float*)d_in[10];
    const float* gq   = (const float*)d_in[11];
    const float* gk   = (const float*)d_in[12];
    const float* gv   = (const float*)d_in[13];
    const float* gg   = (const float*)d_in[14];
    const float* ga_w = (const float*)d_in[15];
    const float* ga_b = (const float*)d_in[16];
    const float* gb_w = (const float*)d_in[17];
    const float* gb_b = (const float*)d_in[18];
    const float* gc_w = (const float*)d_in[19];
    const float* gc_b = (const float*)d_in[20];
    const float* clsw = (const float*)d_in[21];
    float* out = (float*)d_out;

    char* ws = (char*)d_ws;
    size_t off = 0;
    auto alloc = [&](size_t bytes) -> void* {
        void* p = ws + off;
        off += (bytes + 255) & ~size_t(255);
        return p;
    };
    const long XE = (long)NTOK * EMBED;

    bf16_t* xbf = (bf16_t*)alloc(XE * 2);
    bf16_t* wlq = (bf16_t*)alloc(EMBED * EMBED * 2);
    bf16_t* wlk = (bf16_t*)alloc(EMBED * EMBED * 2);
    bf16_t* wlv = (bf16_t*)alloc(EMBED * EMBED * 2);
    bf16_t* wlg = (bf16_t*)alloc(EMBED * EMBED * 2);
    bf16_t* wla = (bf16_t*)alloc(DPOOL * EMBED * 2);
    bf16_t* wlb = (bf16_t*)alloc(DPOOL * EMBED * 2);
    bf16_t* wgq = (bf16_t*)alloc(EMBED * EMBED * 2);
    bf16_t* wgk = (bf16_t*)alloc(EMBED * EMBED * 2);
    bf16_t* wgv = (bf16_t*)alloc(EMBED * EMBED * 2);
    bf16_t* wgg = (bf16_t*)alloc(EMBED * EMBED * 2);
    bf16_t* wga = (bf16_t*)alloc(DPOOL * EMBED * 2);
    bf16_t* wgb = (bf16_t*)alloc(DPOOL * EMBED * 2);
    bf16_t* qbf = (bf16_t*)alloc(XE * 2);
    bf16_t* kbf = (bf16_t*)alloc(XE * 2);
    bf16_t* vbf = (bf16_t*)alloc(XE * 2);
    bf16_t* gbf = (bf16_t*)alloc(XE * 2);
    bf16_t* qrb = (bf16_t*)alloc(XE * 2);
    bf16_t* krb = (bf16_t*)alloc(XE * 2);
    bf16_t* vtb = (bf16_t*)alloc(XE * 2);
    float*  att = (float*)alloc(XE * 4);
    float*  loc = (float*)alloc(XE * 4);
    bf16_t* locb = (bf16_t*)alloc(XE * 2);
    float*  ap  = (float*)alloc((long)NTOK * DPOOL * 4);
    float*  bp  = (float*)alloc((long)NTOK * DPOOL * 4);
    float*  scr = (float*)alloc(NTOK * 4);
    float*  fe  = (float*)alloc(NCH * EMBED * 4);
    bf16_t* feb = (bf16_t*)alloc(NCH * EMBED * 2);
    const long GE = (long)TGLB * EMBED;
    bf16_t* g_q  = (bf16_t*)alloc(GE * 2);
    bf16_t* g_k  = (bf16_t*)alloc(GE * 2);
    bf16_t* g_v  = (bf16_t*)alloc(GE * 2);
    bf16_t* g_g  = (bf16_t*)alloc(GE * 2);
    bf16_t* g_qr = (bf16_t*)alloc(GE * 2);
    bf16_t* g_kr = (bf16_t*)alloc(GE * 2);
    bf16_t* g_vt = (bf16_t*)alloc(GE * 2);
    float*  g_at = (float*)alloc(GE * 4);
    float*  g_lo = (float*)alloc(GE * 4);
    bf16_t* g_lb = (bf16_t*)alloc(GE * 2);
    float*  g_ap = (float*)alloc((long)TGLB * DPOOL * 4);
    float*  g_bp = (float*)alloc((long)TGLB * DPOOL * 4);
    float*  g_sc = (float*)alloc(TGLB * 4);
    float*  fe2  = (float*)alloc(EMBED * 4);

    auto conv = [&](const float* s, bf16_t* dgt, long n) {
        k_cvt_bf16<<<dim3((unsigned)((n + 255) / 256)), dim3(256), 0, stream>>>(s, dgt, n);
    };
    auto gemm = [&](const bf16_t* A, const bf16_t* B, const float* bias,
                    float* Cf, bf16_t* Cb, int M, int N, int K) {
        dim3 grid(N / 64, (M + 63) / 64);
        k_gemm<<<grid, dim3(128), 0, stream>>>(A, B, bias, Cf, Cb, M, N, K);
    };

    // ---- convert inputs/weights to bf16
    conv(x, xbf, XE);
    conv(lq, wlq, EMBED * EMBED); conv(lk, wlk, EMBED * EMBED);
    conv(lv, wlv, EMBED * EMBED); conv(lg, wlg, EMBED * EMBED);
    conv(la_w, wla, DPOOL * EMBED); conv(lb_w, wlb, DPOOL * EMBED);
    conv(gq, wgq, EMBED * EMBED); conv(gk, wgk, EMBED * EMBED);
    conv(gv, wgv, EMBED * EMBED); conv(gg, wgg, EMBED * EMBED);
    conv(ga_w, wga, DPOOL * EMBED); conv(gb_w, wgb, DPOOL * EMBED);

    const float kscale = 1.f / sqrtf((float)EMBED);

    // ---- local retention over 32 chunks of 512 tokens
    gemm(xbf, wlq, nullptr, nullptr, qbf, NTOK, EMBED, EMBED);
    gemm(xbf, wlk, nullptr, nullptr, kbf, NTOK, EMBED, EMBED);
    gemm(xbf, wlv, nullptr, nullptr, vbf, NTOK, EMBED, EMBED);
    gemm(xbf, wlg, nullptr, nullptr, gbf, NTOK, EMBED, EMBED);
    {
        int pairs = NTOK * EMBED / 2;
        k_rope_prep<<<dim3((pairs + 255) / 256), dim3(256), 0, stream>>>(
            qbf, kbf, vbf, qrb, krb, vtb, TLOC, pairs, kscale);
        k_retention<<<dim3(TLOC / 16 / 4, HEADS, NCH), dim3(128), 0, stream>>>(
            qrb, krb, vtb, att, TLOC);
        k_rms_silu<<<dim3((NTOK * HEADS + 255) / 256), dim3(256), 0, stream>>>(
            att, gbf, loc, locb, NTOK);
    }
    // ---- local attention pooling -> fe (32 x 512)
    gemm(locb, wla, la_b, ap, nullptr, NTOK, DPOOL, EMBED);
    gemm(locb, wlb, lb_b, bp, nullptr, NTOK, DPOOL, EMBED);
    k_pool_scores<<<dim3((NTOK + 3) / 4), dim3(128), 0, stream>>>(ap, bp, lc_w, lc_b, scr, NTOK);
    k_pool_weighted<<<dim3(NCH), dim3(256), 0, stream>>>(scr, loc, fe, TLOC);
    conv(fe, feb, NCH * EMBED);

    // ---- global retention over the 32 chunk features
    gemm(feb, wgq, nullptr, nullptr, g_q, TGLB, EMBED, EMBED);
    gemm(feb, wgk, nullptr, nullptr, g_k, TGLB, EMBED, EMBED);
    gemm(feb, wgv, nullptr, nullptr, g_v, TGLB, EMBED, EMBED);
    gemm(feb, wgg, nullptr, nullptr, g_g, TGLB, EMBED, EMBED);
    {
        int pairs = TGLB * EMBED / 2;
        k_rope_prep<<<dim3((pairs + 255) / 256), dim3(256), 0, stream>>>(
            g_q, g_k, g_v, g_qr, g_kr, g_vt, TGLB, pairs, kscale);
        k_retention<<<dim3(1, HEADS, 1), dim3(128), 0, stream>>>(g_qr, g_kr, g_vt, g_at, TGLB);
        k_rms_silu<<<dim3(1), dim3(256), 0, stream>>>(g_at, g_g, g_lo, g_lb, TGLB);
    }
    // ---- global pooling + classifier
    gemm(g_lb, wga, ga_b, g_ap, nullptr, TGLB, DPOOL, EMBED);
    gemm(g_lb, wgb, gb_b, g_bp, nullptr, TGLB, DPOOL, EMBED);
    k_pool_scores<<<dim3((TGLB + 3) / 4), dim3(128), 0, stream>>>(g_ap, g_bp, gc_w, gc_b, g_sc, TGLB);
    k_pool_weighted<<<dim3(1), dim3(256), 0, stream>>>(g_sc, g_lo, fe2, TGLB);
    k_classify<<<dim3(1), dim3(64), 0, stream>>>(fe2, clsw, out);
}
